// MoMe_25254407700662
// MI455X (gfx1250) — matrix operand order repaired and checked by reading, not compiled, not run on hardware
//
#include <hip/hip_runtime.h>

#define T_TOK 65536
#define HDIM  512
#define FFDIM 512
#define NEXP  8
#define PATCH 16

typedef __attribute__((ext_vector_type(16))) _Float16 v16h;
typedef __attribute__((ext_vector_type(8)))  _Float16 v8h;
typedef __attribute__((ext_vector_type(8)))  float    v8f;

// ---------------------------------------------------------------------------
// fp32 -> fp16 elementwise (expert weights)
// ---------------------------------------------------------------------------
__global__ void mome_cvt16(const float* __restrict__ src, _Float16* __restrict__ dst, int n) {
    int i = blockIdx.x * blockDim.x + threadIdx.x;
    if (i < n) dst[i] = (_Float16)src[i];
}

// ---------------------------------------------------------------------------
// patchify + linear projection: tokens[t,d] = sum_l x[t*16+l] * Wp[d,l] + bp[d]
// (K=16, ~1 GFLOP total -> plain FMA; output stored f16 for the WMMA stages)
// ---------------------------------------------------------------------------
__global__ void mome_patchify(const float* __restrict__ x, const float* __restrict__ Wp,
                              const float* __restrict__ bp, _Float16* __restrict__ tok) {
    int idx = blockIdx.x * blockDim.x + threadIdx.x;   // t*H + d
    int t = idx >> 9;
    int d = idx & (HDIM - 1);
    const float* px = x + (size_t)t * PATCH;
    const float* pw = Wp + (size_t)d * PATCH;
    float acc = bp[d];
#pragma unroll
    for (int l = 0; l < PATCH; ++l) acc += px[l] * pw[l];
    tok[idx] = (_Float16)acc;
}

// ---------------------------------------------------------------------------
// router: logits -> fp32 softmax -> top-2 -> dense [T,8] combine weights
// ---------------------------------------------------------------------------
__global__ void mome_router(const _Float16* __restrict__ tok, const float* __restrict__ Wg,
                            float* __restrict__ wbuf) {
    int t = blockIdx.x * blockDim.x + threadIdx.x;
    const _Float16* xr = tok + (size_t)t * HDIM;
    float acc[NEXP];
#pragma unroll
    for (int e = 0; e < NEXP; ++e) acc[e] = 0.f;
    for (int h = 0; h < HDIM; ++h) {
        float xv = (float)xr[h];
#pragma unroll
        for (int e = 0; e < NEXP; ++e) acc[e] += xv * Wg[e * HDIM + h];
    }
    float mx = acc[0];
#pragma unroll
    for (int e = 1; e < NEXP; ++e) mx = fmaxf(mx, acc[e]);
    float sum = 0.f;
#pragma unroll
    for (int e = 0; e < NEXP; ++e) { acc[e] = __expf(acc[e] - mx); sum += acc[e]; }
    float inv = 1.f / sum;
#pragma unroll
    for (int e = 0; e < NEXP; ++e) acc[e] *= inv;
    // top-2, first occurrence wins ties (matches lax.top_k)
    int i1 = 0;
#pragma unroll
    for (int e = 1; e < NEXP; ++e) if (acc[e] > acc[i1]) i1 = e;
    int i2 = (i1 == 0) ? 1 : 0;
#pragma unroll
    for (int e = 0; e < NEXP; ++e) if (e != i1 && acc[e] > acc[i2]) i2 = e;
#pragma unroll
    for (int e = 0; e < NEXP; ++e) {
        float o = 0.f;
        if (e == i1) o = acc[e];
        if (e == i2) o += acc[e];
        wbuf[(size_t)t * NEXP + e] = o;
    }
}

// ---------------------------------------------------------------------------
// MoE FFN: per 32-token tile, loop all 8 experts.
//   Hact = silu(X Wgate^T) * (X Wup^T) * w_e      (GEMM1, f16 -> LDS)
//   out += Hact Wdown^T                           (GEMM2, accumulated in WMMA C regs)
// 8 waves: wave owns m-half (16 rows) x 8 column tiles (128 cols).
// ---------------------------------------------------------------------------
__global__ __launch_bounds__(256)
void mome_ffn(const _Float16* __restrict__ tok, const float* __restrict__ wbuf,
              const _Float16* __restrict__ Wg16, const _Float16* __restrict__ Wu16,
              const _Float16* __restrict__ Wd16, float* __restrict__ out) {
    __shared__ __align__(32) _Float16 Xs[32 * HDIM];   // 32 KB
    __shared__ __align__(32) _Float16 Hs[32 * FFDIM];  // 32 KB

    const int tid  = threadIdx.x;
    const int lane = tid & 31;
    const int wid  = tid >> 5;
    const int t0   = blockIdx.x * 32;

    // ---- stage X tile (32x512 f16) via CDNA5 async global->LDS DMA path ----
    {
        const _Float16* gsrc = tok + (size_t)t0 * HDIM;
        unsigned lds_base = (unsigned)(uintptr_t)(&Xs[0]);
#pragma unroll
        for (int it = 0; it < 8; ++it) {
            int i = tid + it * 256;                  // 2048 x 16B chunks total
            unsigned ldst = lds_base + (unsigned)i * 16u;
            const void* src = (const void*)(gsrc + i * 8);
            asm volatile("global_load_async_to_lds_b128 %0, %1, off"
                         :: "v"(ldst), "v"(src) : "memory");
        }
        asm volatile("s_wait_asynccnt 0" ::: "memory");
    }
    __syncthreads();

    const int mbase  = (wid & 1) * 16;   // m-half of the 32-token tile
    const int ngroup = wid >> 1;         // 0..3, each covers 8 column tiles
    const int mlo    = lane & 15;        // A: row-in-tile, B: col-in-tile
    const int hsel   = lane >> 4;        // lane-half select
    const int kbA    = hsel * 16;        // A frag: 16 contiguous k at kbA
    const int kbB    = hsel * 16;        // B frag: 16 contiguous k at kbB
    const int rbase  = mbase + hsel * 8; // C frag row base (rows rbase..rbase+7)

    v8f oacc[8];
#pragma unroll
    for (int j = 0; j < 8; ++j) oacc[j] = (v8f){};

#pragma unroll 1
    for (int e = 0; e < NEXP; ++e) {
        const _Float16* Bg = Wg16 + (size_t)e * FFDIM * HDIM;
        const _Float16* Bu = Wu16 + (size_t)e * FFDIM * HDIM;
        const _Float16* Bd = Wd16 + (size_t)e * HDIM * FFDIM;

        float wv[8];
#pragma unroll
        for (int r = 0; r < 8; ++r)
            wv[r] = wbuf[(size_t)(t0 + rbase + r) * NEXP + e];

        // ---- GEMM1: gate/up + silu epilogue, router weight folded in ----
#pragma unroll 1
        for (int j = 0; j < 8; ++j) {
            const int ncol = (ngroup * 8 + j) * 16 + mlo;
            v8f ag = (v8f){};
            v8f au = (v8f){};
#pragma unroll 4
            for (int kk = 0; kk < HDIM / 32; ++kk) {
                const int k0 = kk * 32;
                v8h alo = *(const v8h*)&Xs[(mbase + mlo) * HDIM + k0 + kbA];
                v8h ahi = *(const v8h*)&Xs[(mbase + mlo) * HDIM + k0 + kbA + 8];
                v16h a = __builtin_shufflevector(alo, ahi,
                         0,1,2,3,4,5,6,7,8,9,10,11,12,13,14,15);
                v16h bg = *(const v16h*)&Bg[(size_t)ncol * HDIM + k0 + kbB];
                v16h bu = *(const v16h*)&Bu[(size_t)ncol * HDIM + k0 + kbB];
                ag = __builtin_amdgcn_wmma_f32_16x16x32_f16(false, a, false, bg,
                                                            (short)0, ag, false, false);
                au = __builtin_amdgcn_wmma_f32_16x16x32_f16(false, a, false, bu,
                                                            (short)0, au, false, false);
            }
#pragma unroll
            for (int r = 0; r < 8; ++r) {
                float g = ag[r];
                float u = au[r];
                // silu via hardware v_rcp_f32 (TRANS op, co-issues with WMMA)
                float s = g * __builtin_amdgcn_rcpf(1.f + __expf(-g));
                Hs[(rbase + r) * FFDIM + ncol] = (_Float16)(s * u * wv[r]);
            }
        }
        __syncthreads();

        // ---- GEMM2: accumulate into persistent C registers ----
#pragma unroll
        for (int j = 0; j < 8; ++j) {
            const int ncol = (ngroup * 8 + j) * 16 + mlo;
#pragma unroll 4
            for (int kk = 0; kk < FFDIM / 32; ++kk) {
                const int k0 = kk * 32;
                v8h alo = *(const v8h*)&Hs[(mbase + mlo) * FFDIM + k0 + kbA];
                v8h ahi = *(const v8h*)&Hs[(mbase + mlo) * FFDIM + k0 + kbA + 8];
                v16h a = __builtin_shufflevector(alo, ahi,
                         0,1,2,3,4,5,6,7,8,9,10,11,12,13,14,15);
                v16h bd = *(const v16h*)&Bd[(size_t)ncol * FFDIM + k0 + kbB];
                oacc[j] = __builtin_amdgcn_wmma_f32_16x16x32_f16(false, a, false, bd,
                                                                 (short)0, oacc[j], false, false);
            }
        }
        __syncthreads();
    }

    // ---- write output tile (fp32) ----
#pragma unroll
    for (int j = 0; j < 8; ++j) {
        const int ncol = (ngroup * 8 + j) * 16 + mlo;
#pragma unroll
        for (int r = 0; r < 8; ++r) {
            out[(size_t)(t0 + rbase + r) * HDIM + ncol] = oacc[j][r];
        }
    }
}

// ---------------------------------------------------------------------------
extern "C" void kernel_launch(void* const* d_in, const int* in_sizes, int n_in,
                              void* d_out, int out_size, void* d_ws, size_t ws_size,
                              hipStream_t stream) {
    const float* x     = (const float*)d_in[0];
    const float* Wp    = (const float*)d_in[1];
    const float* bp    = (const float*)d_in[2];
    const float* Wg    = (const float*)d_in[3];
    const float* Wgate = (const float*)d_in[4];
    const float* Wup   = (const float*)d_in[5];
    const float* Wdown = (const float*)d_in[6];
    float* out = (float*)d_out;

    // workspace carve-up (~78 MB)
    char* wsb = (char*)d_ws;
    _Float16* tok  = (_Float16*)wsb;                                 // 67,108,864 B
    float*    wbuf = (float*)(wsb + (size_t)67108864);               //  2,097,152 B
    _Float16* Wg16 = (_Float16*)(wsb + (size_t)69206016);            //  4,194,304 B
    _Float16* Wu16 = (_Float16*)(wsb + (size_t)73400320);            //  4,194,304 B
    _Float16* Wd16 = (_Float16*)(wsb + (size_t)77594624);            //  4,194,304 B

    const int nW = NEXP * FFDIM * HDIM;  // 2,097,152 per tensor
    mome_cvt16<<<(nW + 255) / 256, 256, 0, stream>>>(Wgate, Wg16, nW);
    mome_cvt16<<<(nW + 255) / 256, 256, 0, stream>>>(Wup,   Wu16, nW);
    mome_cvt16<<<(nW + 255) / 256, 256, 0, stream>>>(Wdown, Wd16, nW);

    mome_patchify<<<(T_TOK * HDIM) / 256, 256, 0, stream>>>(x, Wp, bp, tok);
    mome_router<<<T_TOK / 256, 256, 0, stream>>>(tok, Wg, wbuf);
    mome_ffn<<<T_TOK / 32, 256, 0, stream>>>(tok, wbuf, Wg16, Wu16, Wd16, out);
}
